// TwoDTPERoPEAttention_28097676050531
// MI455X (gfx1250) — matrix operand order repaired and checked by reading, not compiled, not run on hardware
//
#include <hip/hip_runtime.h>
#include <cstdint>

#define D_MODEL 768
#define N_HEADS 12
#define HEAD_DIM 64
#define L_SEQ 1024
#define B_SZ 8
#define NTOK (B_SZ * L_SEQ)              /* 8192 token rows           */
#define NTH  (NTOK * N_HEADS)            /* 98304 token-heads         */
#define BHLD ((size_t)B_SZ * N_HEADS * L_SEQ * HEAD_DIM) /* 6291456  */

typedef __attribute__((ext_vector_type(16))) _Float16 v16h;
typedef __attribute__((ext_vector_type(8)))  float    v8f;

__device__ __forceinline__ v8f zero8() {
  v8f z = {0.f, 0.f, 0.f, 0.f, 0.f, 0.f, 0.f, 0.f};
  return z;
}
__device__ __forceinline__ v8f wmma16(v16h a, v16h b, v8f c) {
  return __builtin_amdgcn_wmma_f32_16x16x32_f16(false, a, false, b,
                                                (short)0, c, false, false);
}
// Load a 16-half fragment from two 8-half (16B) runs.
__device__ __forceinline__ v16h ldfrag2(const _Float16* p0, const _Float16* p1) {
  v16h r;
  reinterpret_cast<uint4*>(&r)[0] = *reinterpret_cast<const uint4*>(p0);
  reinterpret_cast<uint4*>(&r)[1] = *reinterpret_cast<const uint4*>(p1);
  return r;
}
__device__ __forceinline__ v16h ldfrag(const _Float16* p) { return ldfrag2(p, p + 8); }

// ---------------------------------------------------------------- utilities
__global__ void rope_tables_kernel(float* __restrict__ cost,
                                   float* __restrict__ sint,
                                   float* __restrict__ entacc) {
  int p = blockIdx.x, i = threadIdx.x;             // 512 x 32
  float inv = __expf(-(float)i * (9.210340371976184f / 32.0f)); // 10000^(-i/32)
  float a = (float)p * inv;
  cost[p * 32 + i] = cosf(a);
  sint[p * 32 + i] = sinf(a);
  if (p == 0 && i == 0) *entacc = 0.0f;
}

__global__ void cvt_f32_f16(const float* __restrict__ s, _Float16* __restrict__ d, int n) {
  int i = blockIdx.x * blockDim.x + threadIdx.x;
  if (i < n) d[i] = (_Float16)s[i];
}

// -------------------------------------------------- GEMM: C = A * W^T + bias
// A: M x K (f16, row-major), W: N x K (f16, row-major), C: M x N (f32)
// M%64==0, N%128==0, K%64==0. 8 waves, each wave does a 32x32 (2x2 WMMA) tile.
// BK=64, double-buffered LDS: one barrier per K-step; next tile's global loads
// (clamped index, unconditional -> no scratch demotion) issued before the 8
// WMMAs; staging kept in named uint4 scalars so they stay in VGPRs.
__global__ __launch_bounds__(256) void gemm_xwt_wmma(
    const _Float16* __restrict__ A, const _Float16* __restrict__ W,
    const float* __restrict__ bias, float* __restrict__ C,
    int M, int N, int K) {
  __shared__ _Float16 As[2][64 * 72];
  __shared__ _Float16 Bs[2][128 * 72];
  const int t = threadIdx.x;
  const int lane = t & 31, wv = t >> 5;
  const int g = lane >> 4, m = lane & 15;
  const int wm = wv >> 2, wn = wv & 3;            // waves: 2 (M) x 4 (N)
  const int row0 = blockIdx.x * 64;
  const int col0 = blockIdx.y * 128;

  // staging assignment: A 64x64 halves (16/thr), B 128x64 halves (32/thr)
  const int ar = t >> 2, ac = (t & 3) * 16;
  const int br = t >> 1, bc = (t & 1) * 32;
  const _Float16* gA = &A[(size_t)(row0 + ar) * K + ac];
  const _Float16* gW = &W[(size_t)(col0 + br) * K + bc];

  v8f acc00 = zero8(), acc01 = zero8(), acc10 = zero8(), acc11 = zero8();

  { // preload tile 0
    const uint4* pa = reinterpret_cast<const uint4*>(gA);
    const uint4* pw = reinterpret_cast<const uint4*>(gW);
    uint4 a0 = pa[0], a1 = pa[1];
    uint4 b0 = pw[0], b1 = pw[1], b2 = pw[2], b3 = pw[3];
    uint4* da = reinterpret_cast<uint4*>(&As[0][ar * 72 + ac]);
    uint4* db = reinterpret_cast<uint4*>(&Bs[0][br * 72 + bc]);
    da[0] = a0; da[1] = a1;
    db[0] = b0; db[1] = b1; db[2] = b2; db[3] = b3;
  }
  __syncthreads();

  const int nk = K >> 6;
  int buf = 0;
  for (int ks = 0; ks < nk; ++ks) {
    // unconditionally issue loads for the (clamped) next tile
    const int knext = (ks + 1 < nk) ? (ks + 1) : ks;
    const uint4* pa = reinterpret_cast<const uint4*>(gA + knext * 64);
    const uint4* pw = reinterpret_cast<const uint4*>(gW + knext * 64);
    uint4 an0 = pa[0], an1 = pa[1];
    uint4 bn0 = pw[0], bn1 = pw[1], bn2 = pw[2], bn3 = pw[3];
    if (ks + 2 < nk) {                 // gfx1250 global_prefetch for tile+2
      __builtin_prefetch(gA + (ks + 2) * 64, 0, 1);
      __builtin_prefetch(gW + (ks + 2) * 64, 0, 1);
    }
    // compute current tile from LDS[buf]
    const _Float16* as = As[buf];
    const _Float16* bs = Bs[buf];
    for (int kc = 0; kc < 2; ++kc) {
      v16h af0, af1, bf0, bf1;
      {
        int r0 = (wm * 32 + m) * 72 + kc * 32;
        int r1 = (wm * 32 + 16 + m) * 72 + kc * 32;
        af0 = ldfrag2(&as[r0 + 8 * g], &as[r0 + 16 + 8 * g]); // A: K=8g+j / 16+8g+j
        af1 = ldfrag2(&as[r1 + 8 * g], &as[r1 + 16 + 8 * g]);
        int c0 = (wn * 32 + m) * 72 + kc * 32;
        int c1 = (wn * 32 + 16 + m) * 72 + kc * 32;
        bf0 = ldfrag2(&bs[c0 + 8 * g], &bs[c0 + 16 + 8 * g]); // B: K=8g+j / 16+8g+j
        bf1 = ldfrag2(&bs[c1 + 8 * g], &bs[c1 + 16 + 8 * g]);
      }
      acc00 = wmma16(af0, bf0, acc00);
      acc01 = wmma16(af0, bf1, acc01);
      acc10 = wmma16(af1, bf0, acc10);
      acc11 = wmma16(af1, bf1, acc11);
    }
    if (ks + 1 < nk) {                 // stage next tile into the other buffer
      int nb = buf ^ 1;
      uint4* da = reinterpret_cast<uint4*>(&As[nb][ar * 72 + ac]);
      uint4* db = reinterpret_cast<uint4*>(&Bs[nb][br * 72 + bc]);
      da[0] = an0; da[1] = an1;
      db[0] = bn0; db[1] = bn1; db[2] = bn2; db[3] = bn3;
      __syncthreads();
      buf = nb;
    }
  }

  for (int r = 0; r < 8; ++r) {
    int rowA = row0 + wm * 32 + 8 * g + r;                    // C layout
    int rowB = rowA + 16;
    int colA = col0 + wn * 32 + m;
    int colB = colA + 16;
    C[(size_t)rowA * N + colA] = acc00[r] + bias[colA];
    C[(size_t)rowA * N + colB] = acc01[r] + bias[colB];
    C[(size_t)rowB * N + colA] = acc10[r] + bias[colA];
    C[(size_t)rowB * N + colB] = acc11[r] + bias[colB];
  }
}

// --------------------------------------------- RoPE (interleaved pairs) + cvt
__global__ __launch_bounds__(256) void rope_apply_kernel(
    const float* __restrict__ Qf, const float* __restrict__ Kf, const float* __restrict__ Vf,
    const int* __restrict__ pos_row, const int* __restrict__ pos_col,
    const float* __restrict__ cost, const float* __restrict__ sint,
    _Float16* __restrict__ qr, _Float16* __restrict__ kr,
    _Float16* __restrict__ qc, _Float16* __restrict__ kc,
    _Float16* __restrict__ vh) {
  int idx = blockIdx.x * 256 + threadIdx.x;   // (b*L + l)*H + h
  int bl = idx / N_HEADS, h = idx % N_HEADS;
  int b = bl / L_SEQ, l = bl % L_SEQ;
  int pr = pos_row[bl], pc = pos_col[bl];
  size_t s = (size_t)bl * D_MODEL + h * HEAD_DIM;
  size_t d = (((size_t)b * N_HEADS + h) * L_SEQ + l) * HEAD_DIM;
  const float sc = 0.125f;                    // 1/sqrt(64) folded into q
  for (int i = 0; i < 32; ++i) {
    float c0 = cost[pr * 32 + i], s0 = sint[pr * 32 + i];
    float c1 = cost[pc * 32 + i], s1 = sint[pc * 32 + i];
    float q0 = Qf[s + 2 * i], q1 = Qf[s + 2 * i + 1];
    float k0 = Kf[s + 2 * i], k1 = Kf[s + 2 * i + 1];
    qr[d + 2 * i]     = (_Float16)((q0 * c0 - q1 * s0) * sc);
    qr[d + 2 * i + 1] = (_Float16)((q1 * c0 + q0 * s0) * sc);
    kr[d + 2 * i]     = (_Float16)(k0 * c0 - k1 * s0);
    kr[d + 2 * i + 1] = (_Float16)(k1 * c0 + k0 * s0);
    qc[d + 2 * i]     = (_Float16)((q0 * c1 - q1 * s1) * sc);
    qc[d + 2 * i + 1] = (_Float16)((q1 * c1 + q0 * s1) * sc);
    kc[d + 2 * i]     = (_Float16)(k0 * c1 - k1 * s1);
    kc[d + 2 * i + 1] = (_Float16)(k1 * c1 + k0 * s1);
    vh[d + 2 * i]     = (_Float16)Vf[s + 2 * i];
    vh[d + 2 * i + 1] = (_Float16)Vf[s + 2 * i + 1];
  }
}

// ------------------------------------------------ causal flash attention (WMMA)
// grid: (B*H, L/64, 2). 4 waves x 16 query rows. Key blocks of 32.
__global__ __launch_bounds__(128) void attn_flash_kernel(
    const _Float16* __restrict__ qrowP, const _Float16* __restrict__ krowP,
    const _Float16* __restrict__ qcolP, const _Float16* __restrict__ kcolP,
    const _Float16* __restrict__ vhP,
    float* __restrict__ orowP, float* __restrict__ ocolP) {
  const int bh = blockIdx.x;
  const int qb = blockIdx.y;
  const int z  = blockIdx.z;
  const _Float16* qp = z ? qcolP : qrowP;
  const _Float16* kp = z ? kcolP : krowP;
  float* op = z ? ocolP : orowP;

  __shared__ _Float16 Qs[64 * 72];
  __shared__ _Float16 Ks[32 * 72];
  __shared__ _Float16 Vt[64 * 40];     // V transposed: Vt[d][k]
  __shared__ _Float16 Ps[4 * 16 * 40]; // per-wave P staging (C->A layout)

  const int t = threadIdx.x;
  const int lane = t & 31, w = t >> 5;
  const int g = lane >> 4, m = lane & 15;

  { // stage Q tile 64x64
    int r = t >> 1, c = (t & 1) * 32;
    const uint4* src = reinterpret_cast<const uint4*>(
        &qp[((size_t)bh * L_SEQ + qb * 64 + r) * 64 + c]);
    uint4* dst = reinterpret_cast<uint4*>(&Qs[r * 72 + c]);
    dst[0] = src[0]; dst[1] = src[1]; dst[2] = src[2]; dst[3] = src[3];
  }
  __syncthreads();
  v16h qa0, qa1;                       // Q A-frags for k 0..31 and 32..63
  {
    int r = (w * 16 + m) * 72;
    qa0 = ldfrag2(&Qs[r + 8 * g], &Qs[r + 16 + 8 * g]);
    qa1 = ldfrag2(&Qs[r + 32 + 8 * g], &Qs[r + 48 + 8 * g]);
  }

  float m_i[8], l_i[8];
  v8f oacc[4] = {zero8(), zero8(), zero8(), zero8()};
  for (int r = 0; r < 8; ++r) { m_i[r] = -1e30f; l_i[r] = 0.f; }

  const int nkb = qb * 2 + 2;          // keys up to qb*64+63 (causal)
  for (int kb = 0; kb < nkb; ++kb) {
    __syncthreads();
    { // stage K tile 32x64 row-major
      int r = t >> 2, c = (t & 3) * 16;
      const uint4* src = reinterpret_cast<const uint4*>(
          &kp[((size_t)bh * L_SEQ + kb * 32 + r) * 64 + c]);
      uint4* dst = reinterpret_cast<uint4*>(&Ks[r * 72 + c]);
      dst[0] = src[0]; dst[1] = src[1];
    }
    for (int u = 0; u < 2; ++u) {      // stage V transposed
      int idx = t + u * 128;
      int key = idx & 31, d0 = (idx >> 5) * 8;
      uint4 pack = *reinterpret_cast<const uint4*>(
          &vhP[((size_t)bh * L_SEQ + kb * 32 + key) * 64 + d0]);
      const _Float16* hs = reinterpret_cast<const _Float16*>(&pack);
      for (int i = 0; i < 8; ++i) Vt[(d0 + i) * 40 + key] = hs[i];
    }
    if (kb + 1 < nkb) {                // gfx1250 global_prefetch of next block
      __builtin_prefetch(&kp[((size_t)bh * L_SEQ + (kb + 1) * 32 + (t >> 2)) * 64], 0, 1);
      __builtin_prefetch(&vhP[((size_t)bh * L_SEQ + (kb + 1) * 32 + (t & 31)) * 64], 0, 1);
    }
    __syncthreads();

    // S = Q K^T : two 16x16 subtiles (keys kb*32 + [0,16) and [16,32))
    v8f s0 = zero8(), s1 = zero8();
    {
      v16h b00 = ldfrag2(&Ks[m * 72 + 8 * g], &Ks[m * 72 + 16 + 8 * g]);
      v16h b01 = ldfrag2(&Ks[m * 72 + 32 + 8 * g], &Ks[m * 72 + 48 + 8 * g]);
      s0 = wmma16(qa0, b00, s0);
      s0 = wmma16(qa1, b01, s0);
      v16h b10 = ldfrag2(&Ks[(16 + m) * 72 + 8 * g], &Ks[(16 + m) * 72 + 16 + 8 * g]);
      v16h b11 = ldfrag2(&Ks[(16 + m) * 72 + 32 + 8 * g], &Ks[(16 + m) * 72 + 48 + 8 * g]);
      s1 = wmma16(qa0, b10, s1);
      s1 = wmma16(qa1, b11, s1);
    }

    const int qbase = qb * 64 + w * 16 + 8 * g;
    const int key0 = kb * 32 + m, key1 = kb * 32 + 16 + m;
    for (int r = 0; r < 8; ++r) {      // causal mask
      int q = qbase + r;
      if (key0 > q) s0[r] = -1e30f;
      if (key1 > q) s1[r] = -1e30f;
    }
    for (int r = 0; r < 8; ++r) {      // online softmax per row
      float mx = fmaxf(s0[r], s1[r]);
      for (int msk = 8; msk; msk >>= 1) mx = fmaxf(mx, __shfl_xor(mx, msk, 32));
      float mnew = fmaxf(m_i[r], mx);
      float scale = __expf(m_i[r] - mnew);
      float p0 = __expf(s0[r] - mnew);
      float p1 = __expf(s1[r] - mnew);
      float rs = p0 + p1;
      for (int msk = 8; msk; msk >>= 1) rs += __shfl_xor(rs, msk, 32);
      l_i[r] = l_i[r] * scale + rs;
      m_i[r] = mnew;
      for (int c = 0; c < 4; ++c) oacc[c][r] *= scale;
      Ps[w * 640 + (8 * g + r) * 40 + m]      = (_Float16)p0;
      Ps[w * 640 + (8 * g + r) * 40 + 16 + m] = (_Float16)p1;
    }
    __syncthreads();
    v16h pa;                            // P as 16x32 A-frag
    {
      int r = w * 640 + m * 40;
      pa = ldfrag2(&Ps[r + 8 * g], &Ps[r + 16 + 8 * g]);
    }
    for (int c = 0; c < 4; ++c) {       // O += P * V
      v16h vb = ldfrag2(&Vt[(c * 16 + m) * 40 + 8 * g], &Vt[(c * 16 + m) * 40 + 16 + 8 * g]);
      oacc[c] = wmma16(pa, vb, oacc[c]);
    }
  }
  for (int c = 0; c < 4; ++c)
    for (int r = 0; r < 8; ++r) {
      size_t o = ((size_t)bh * L_SEQ + qb * 64 + w * 16 + 8 * g + r) * 64 + c * 16 + m;
      op[o] = oacc[c][r] / l_i[r];
    }
}

// ------------------------------------------- router + mix + entropy reduction
__global__ __launch_bounds__(256) void router_mix_kernel(
    const float* __restrict__ Qf,
    const float* __restrict__ orow, const float* __restrict__ ocol,
    const float* __restrict__ Wu, const float* __restrict__ bu,
    const float* __restrict__ Wd, const float* __restrict__ bd,
    _Float16* __restrict__ mixedh, float* __restrict__ entacc) {
  __shared__ _Float16 sWu[256 * 64];
  __shared__ float sbu[256], sWd[512];
  int t = threadIdx.x;
  for (int i = t; i < 256 * 64; i += 256) sWu[i] = (_Float16)Wu[i];
  if (t < 256) sbu[t] = bu[t];
  for (int i = t; i < 512; i += 256) sWd[i] = Wd[i];
  __syncthreads();

  int idx = blockIdx.x * 256 + t;      // (b*L + l)*H + h
  int bl = idx / N_HEADS, h = idx % N_HEADS;
  int b = bl / L_SEQ, l = bl % L_SEQ;
  float q[64];
  size_t qs = (size_t)bl * D_MODEL + h * HEAD_DIM;
  for (int d = 0; d < 64; ++d) q[d] = Qf[qs + d];
  float l0 = bd[0], l1 = bd[1];
  for (int e = 0; e < 256; ++e) {
    float a = sbu[e];
    const _Float16* wr = &sWu[e * 64];
    for (int d = 0; d < 64; ++d) a += q[d] * (float)wr[d];
    float gg = 0.5f * a * (1.0f + tanhf(0.7978845608028654f * (a + 0.044715f * a * a * a)));
    l0 += gg * sWd[e];
    l1 += gg * sWd[256 + e];
  }
  float mx = fmaxf(l0, l1);
  float e0 = __expf(l0 - mx), e1 = __expf(l1 - mx);
  float inv = 1.0f / (e0 + e1);
  float r0 = e0 * inv, r1 = e1 * inv;
  float ent = -(r0 * __logf(r0 + 1e-9f) + r1 * __logf(r1 + 1e-9f));
  for (int msk = 16; msk; msk >>= 1) ent += __shfl_xor(ent, msk, 32);
  if ((t & 31) == 0) atomicAdd(entacc, ent);
  size_t ao = (((size_t)b * N_HEADS + h) * L_SEQ + l) * HEAD_DIM;
  size_t mo = (size_t)bl * D_MODEL + h * HEAD_DIM;
  for (int d = 0; d < 64; ++d)
    mixedh[mo + d] = (_Float16)(r0 * orow[ao + d] + r1 * ocol[ao + d]);
}

__global__ void finalize_entropy_kernel(const float* __restrict__ entacc,
                                        float* __restrict__ out) {
  out[0] = entacc[0] / (float)NTH;
}

// --------------------------------------------------------------------- launch
extern "C" void kernel_launch(void* const* d_in, const int* in_sizes, int n_in,
                              void* d_out, int out_size, void* d_ws, size_t ws_size,
                              hipStream_t stream) {
  (void)in_sizes; (void)n_in; (void)ws_size;
  const float* hidden  = (const float*)d_in[0];
  const int*   pos_row = (const int*)d_in[1];
  const int*   pos_col = (const int*)d_in[2];
  const float* Wq = (const float*)d_in[3];
  const float* bq = (const float*)d_in[4];
  const float* Wk = (const float*)d_in[5];
  const float* bk = (const float*)d_in[6];
  const float* Wv = (const float*)d_in[7];
  const float* bv = (const float*)d_in[8];
  const float* Wo = (const float*)d_in[9];
  const float* bo = (const float*)d_in[10];
  const float* Wu = (const float*)d_in[11];
  const float* bu = (const float*)d_in[12];
  const float* Wd = (const float*)d_in[13];
  const float* bd = (const float*)d_in[14];

  char* ws = (char*)d_ws;
  size_t off = 0;
  auto alloc = [&](size_t bytes) -> void* {
    void* p = ws + off;
    off = (off + bytes + 255) & ~(size_t)255;
    return p;
  };
  float*    cost    = (float*)alloc(512 * 32 * sizeof(float));
  float*    sint    = (float*)alloc(512 * 32 * sizeof(float));
  float*    entacc  = (float*)alloc(sizeof(float));
  _Float16* hiddenH = (_Float16*)alloc((size_t)NTOK * D_MODEL * 2);
  _Float16* WqH     = (_Float16*)alloc((size_t)D_MODEL * D_MODEL * 2);
  _Float16* WkH     = (_Float16*)alloc((size_t)D_MODEL * D_MODEL * 2);
  _Float16* WvH     = (_Float16*)alloc((size_t)D_MODEL * D_MODEL * 2);
  _Float16* WoH     = (_Float16*)alloc((size_t)D_MODEL * D_MODEL * 2);
  float*    Qf      = (float*)alloc((size_t)NTOK * D_MODEL * 4);
  float*    Kf      = (float*)alloc((size_t)NTOK * D_MODEL * 4);
  float*    Vf      = (float*)alloc((size_t)NTOK * D_MODEL * 4);
  _Float16* qrowB   = (_Float16*)alloc(BHLD * 2);
  _Float16* krowB   = (_Float16*)alloc(BHLD * 2);
  _Float16* qcolB   = (_Float16*)alloc(BHLD * 2);
  _Float16* kcolB   = (_Float16*)alloc(BHLD * 2);
  _Float16* vhB     = (_Float16*)alloc(BHLD * 2);
  float*    orowB   = (float*)alloc(BHLD * 4);
  float*    ocolB   = (float*)alloc(BHLD * 4);
  _Float16* mixedH  = (_Float16*)alloc((size_t)NTOK * D_MODEL * 2);

  const int nHidden = NTOK * D_MODEL;      // 6291456
  const int nW      = D_MODEL * D_MODEL;   // 589824

  rope_tables_kernel<<<512, 32, 0, stream>>>(cost, sint, entacc);
  cvt_f32_f16<<<(nHidden + 255) / 256, 256, 0, stream>>>(hidden, hiddenH, nHidden);
  cvt_f32_f16<<<(nW + 255) / 256, 256, 0, stream>>>(Wq, WqH, nW);
  cvt_f32_f16<<<(nW + 255) / 256, 256, 0, stream>>>(Wk, WkH, nW);
  cvt_f32_f16<<<(nW + 255) / 256, 256, 0, stream>>>(Wv, WvH, nW);
  cvt_f32_f16<<<(nW + 255) / 256, 256, 0, stream>>>(Wo, WoH, nW);

  dim3 ggrid(NTOK / 64, D_MODEL / 128);    // 128 x 6
  gemm_xwt_wmma<<<ggrid, 256, 0, stream>>>(hiddenH, WqH, bq, Qf, NTOK, D_MODEL, D_MODEL);
  gemm_xwt_wmma<<<ggrid, 256, 0, stream>>>(hiddenH, WkH, bk, Kf, NTOK, D_MODEL, D_MODEL);
  gemm_xwt_wmma<<<ggrid, 256, 0, stream>>>(hiddenH, WvH, bv, Vf, NTOK, D_MODEL, D_MODEL);

  rope_apply_kernel<<<NTH / 256, 256, 0, stream>>>(Qf, Kf, Vf, pos_row, pos_col,
                                                   cost, sint, qrowB, krowB, qcolB,
                                                   kcolB, vhB);

  attn_flash_kernel<<<dim3(B_SZ * N_HEADS, L_SEQ / 64, 2), 128, 0, stream>>>(
      qrowB, krowB, qcolB, kcolB, vhB, orowB, ocolB);

  router_mix_kernel<<<NTH / 256, 256, 0, stream>>>(Qf, orowB, ocolB, Wu, bu, Wd, bd,
                                                   mixedH, entacc);

  gemm_xwt_wmma<<<ggrid, 256, 0, stream>>>(mixedH, WoH, bo, (float*)d_out,
                                           NTOK, D_MODEL, D_MODEL);

  finalize_entropy_kernel<<<1, 1, 0, stream>>>(entacc, (float*)d_out + (out_size - 1));
}
